// HSMSSD3D_49555332661307
// MI455X (gfx1250) — compile-verified, matrix operand order checked by reading, not run on hardware
//
#include <hip/hip_runtime.h>
#include <hip/hip_bf16.h>
#include <stdint.h>

// HSMSSD3D for MI455X (gfx1250, wave32, WMMA).
// Memory-bound workload (~1.5 GB traffic vs 63 GFLOP): big GEMMs use
// v_wmma_f32_16x16x32_bf16 with f32 accumulation. Small reused operands
// (W_bcdt, h4, AB, C_) are pre-converted to bf16 in workspace so WMMA
// fragments are loaded as aligned 32B chunks and bit-cast straight into the
// packed VGPR layout (no per-element conversion VALU in the hot loops).

#define SS   64
#define DM   256
#define NN   65536          // 16*64*64
#define BB   4

typedef __attribute__((ext_vector_type(16))) __bf16 v16bf;
typedef __attribute__((ext_vector_type(8)))  float  v8f;
typedef __attribute__((ext_vector_type(8)))  unsigned v8u;

static __device__ __forceinline__ unsigned short f2bfu(float f) {
  unsigned u = __builtin_bit_cast(unsigned, f);
  return (unsigned short)((u + 0x7FFFu + ((u >> 16) & 1u)) >> 16); // RNE
}
static __device__ __forceinline__ unsigned pk2bf(float a, float b) {
  return (unsigned)f2bfu(a) | ((unsigned)f2bfu(b) << 16);
}
static __device__ __forceinline__ __bf16 us2bf(unsigned short s) {
  return __builtin_bit_cast(__bf16, s);
}
// 16 contiguous f32 -> bf16 fragment (lane layout: pairs (2v,2v+1) per dword)
static __device__ __forceinline__ v16bf load16f_bf(const float* __restrict__ p) {
  v8u t;
#pragma unroll
  for (int v = 0; v < 8; ++v) t[v] = pk2bf(p[2 * v], p[2 * v + 1]);
  return __builtin_bit_cast(v16bf, t);
}
// 16 contiguous, 32B-aligned bf16 -> fragment: pure load, zero VALU
static __device__ __forceinline__ v16bf loadfrag_bf(const unsigned short* __restrict__ p) {
  return __builtin_bit_cast(v16bf, *(const v8u*)(const void*)p);
}

// ---------------- K0: W_bcdt (192x256) -> bf16 once ------------------------
__global__ void k0_wcvt(const float* __restrict__ Wb,
                        unsigned short* __restrict__ wbf) {
  int idx = blockIdx.x * 256 + threadIdx.x;   // 49152
  wbf[idx] = f2bfu(Wb[idx]);
}

// ---------------- K1: BCdt = W_bcdt(192x256) x X(256xNN) + b  (per batch) ---
// One wave per (b, 16-col N tile). The 256x16 X panel is held resident in
// VGPRs as bf16 (8 k-steps x v16bf = 64 VGPRs) and reused for all 12 M tiles,
// so x is streamed from HBM exactly once. A fragments: 32B bf16 loads.
__global__ void k1_bcdt_gemm(const float* __restrict__ x,
                             const unsigned short* __restrict__ wbf,
                             const float* __restrict__ bbias,
                             float* __restrict__ out) {
  const int lane = threadIdx.x & 31;
  const int wid  = threadIdx.x >> 5;
  const int task = blockIdx.x * 8 + wid;     // 16384 tasks = 4 * 4096
  const int b    = task >> 12;
  const int nt   = task & 4095;
  const long n0  = (long)nt * 16;
  const int ncol = lane & 15;
  const int half = lane >> 4;

  const float* xb = x + (long)b * DM * NN;

  v16bf Bp[8];
#pragma unroll
  for (int ks = 0; ks < 8; ++ks) {
    const float* p = xb + (long)(ks * 32 + 16 * half) * NN + n0 + ncol;
    v8u t;
#pragma unroll
    for (int v = 0; v < 8; ++v)
      t[v] = pk2bf(p[(long)(2 * v) * NN], p[(long)(2 * v + 1) * NN]);
    Bp[ks] = __builtin_bit_cast(v16bf, t);
  }

  for (int mt = 0; mt < 12; ++mt) {
    const int m0 = mt * 16;
    v8f acc;
#pragma unroll
    for (int j = 0; j < 8; ++j) acc[j] = bbias[m0 + j + 8 * half];
#pragma unroll
    for (int ks = 0; ks < 8; ++ks) {
      v16bf A = loadfrag_bf(wbf + (long)(m0 + ncol) * DM + ks * 32 + 16 * half);
      acc = __builtin_amdgcn_wmma_f32_16x16x32_bf16(false, A, false, Bp[ks],
                                                    (short)0, acc, false, false);
    }
    float* orow = out + ((long)b * 192 + m0) * NN + n0 + ncol;
#pragma unroll
    for (int j = 0; j < 8; ++j)
      orow[(long)(j + 8 * half) * NN] = acc[j];
  }
}

// ---------------- K2: depthwise 3x3x3 conv + bias, route channels ----------
// out ch [0,64)   -> B_  (f32, bdt rows 0..63)
// out ch [64,128) -> C_  (bf16, cbf)
// out ch [128,192)-> dt  (f32, bdt rows 64..127)
__global__ void k2_dwconv(const float* __restrict__ inp,
                          const float* __restrict__ Wd,
                          const float* __restrict__ bd,
                          float* __restrict__ bdt,
                          unsigned short* __restrict__ cbf) {
  long idx = (long)blockIdx.x * 256 + threadIdx.x;   // 4*192*NN threads
  int n = (int)(idx & (NN - 1));
  int t = (int)(idx >> 16);
  int o = t % 192;
  int b = t / 192;
  int d = n >> 12, h = (n >> 6) & 63, w = n & 63;

  const float* in = inp + ((long)b * 192 + o) * NN;
  const float* wk = Wd + o * 27;
  float acc = bd[o];
#pragma unroll
  for (int kd = 0; kd < 3; ++kd) {
    int id = d + kd - 1;
    if (id < 0 || id > 15) continue;
#pragma unroll
    for (int kh = 0; kh < 3; ++kh) {
      int ih = h + kh - 1;
      if (ih < 0 || ih > 63) continue;
#pragma unroll
      for (int kw = 0; kw < 3; ++kw) {
        int iw = w + kw - 1;
        if (iw < 0 || iw > 63) continue;
        acc += wk[kd * 9 + kh * 3 + kw] * in[id * 4096 + ih * 64 + iw];
      }
    }
  }
  if (o < 64)        bdt[((long)b * 128 + o) * NN + n] = acc;
  else if (o < 128)  cbf[((long)b * 64 + (o - 64)) * NN + n] = f2bfu(acc);
  else               bdt[((long)b * 128 + (o - 64)) * NN + n] = acc;
}

// ---------------- K3: softmax stats over n (per b,s row of dt) -------------
// softmax(dt + A[s]) over n == softmax(dt) over n (A[s] constant -> cancels).
__global__ void k3_softmax_stats(const float* __restrict__ bdt,
                                 float* __restrict__ stats) {
  __shared__ float sm[256];
  int row = blockIdx.x;               // b*64+s, 256 rows
  int b = row >> 6, s = row & 63;
  const float* dt = bdt + ((long)b * 128 + 64 + s) * NN;
  int t = threadIdx.x;

  float mx = -3.4e38f;
  for (int n = t; n < NN; n += 256) mx = fmaxf(mx, dt[n]);
  sm[t] = mx; __syncthreads();
  for (int off = 128; off > 0; off >>= 1) {
    if (t < off) sm[t] = fmaxf(sm[t], sm[t + off]);
    __syncthreads();
  }
  mx = sm[0]; __syncthreads();

  float sum = 0.f;
  for (int n = t; n < NN; n += 256) sum += __expf(dt[n] - mx);
  sm[t] = sum; __syncthreads();
  for (int off = 128; off > 0; off >>= 1) {
    if (t < off) sm[t] += sm[t + off];
    __syncthreads();
  }
  if (t == 0) { stats[row * 2] = mx; stats[row * 2 + 1] = 1.0f / sm[0]; }
}

// ---------------- K4: AB = softmax(dt) * B_  -> bf16 -----------------------
__global__ void k4_ab(const float* __restrict__ bdt,
                      const float* __restrict__ stats,
                      unsigned short* __restrict__ ab) {
  long idx = (long)blockIdx.x * 256 + threadIdx.x;   // (b*64+s)*NN + n
  int n = (int)(idx & (NN - 1));
  int row = (int)(idx >> 16);
  int b = row >> 6, s = row & 63;
  float mx = stats[row * 2], inv = stats[row * 2 + 1];
  float dt = bdt[((long)b * 128 + 64 + s) * NN + n];
  float Bv = bdt[((long)b * 128 + s) * NN + n];
  ab[idx] = f2bfu(__expf(dt - mx) * inv * Bv);
}

// ---------------- K5: h = x(256xNN) . AB^T (K=NN), split-K WMMA ------------
// Wave task = (b, ctile, chunk): chunk = 1024 of K; each wave computes a
// 16(c) x 64(s) partial (4 accumulators). B fragments are pure 32B bf16
// loads (pair packing matches WMMA layout). No atomics: partials -> K5b.
__global__ void k5_hgemm(const float* __restrict__ x,
                         const unsigned short* __restrict__ ab,
                         float* __restrict__ part) {
  const int lane = threadIdx.x & 31;
  const int wid  = threadIdx.x >> 5;
  const int task = blockIdx.x * 8 + wid;   // 4096 = 4 * 16 * 64
  const int chunk = task & 63;
  const int ct    = (task >> 6) & 15;
  const int b     = task >> 10;
  const int ncol  = lane & 15;
  const int half  = lane >> 4;
  const long k0   = (long)chunk * 1024;

  v8f acc[4];
#pragma unroll
  for (int st = 0; st < 4; ++st)
#pragma unroll
    for (int j = 0; j < 8; ++j) acc[st][j] = 0.f;

  const float* xrow = x + ((long)b * DM + ct * 16 + ncol) * NN + k0 + 16 * half;
  const unsigned short* abb = ab + (long)b * SS * NN;

  for (int kk = 0; kk < 32; ++kk) {
    v16bf A = load16f_bf(xrow + kk * 32);
#pragma unroll
    for (int st = 0; st < 4; ++st) {
      v16bf Bv = loadfrag_bf(abb + (long)(st * 16 + ncol) * NN +
                             k0 + kk * 32 + 16 * half);
      acc[st] = __builtin_amdgcn_wmma_f32_16x16x32_bf16(false, A, false, Bv,
                                                        (short)0, acc[st],
                                                        false, false);
    }
  }
  float* pr = part + ((long)chunk * 4 + b) * (256 * 64);
#pragma unroll
  for (int st = 0; st < 4; ++st)
#pragma unroll
    for (int j = 0; j < 8; ++j)
      pr[(long)(ct * 16 + j + 8 * half) * 64 + st * 16 + ncol] = acc[st][j];
}

// ---------------- K5b: reduce split-K partials -> h ------------------------
__global__ void k5b_reduce(const float* __restrict__ part,
                           float* __restrict__ h) {
  int idx = blockIdx.x * 256 + threadIdx.x;   // (b*256+c)*64+s, 65536
  float s = 0.f;
  for (int ch = 0; ch < 64; ++ch) s += part[(long)ch * 65536 + idx];
  h[idx] = s;
}

// ---------------- K6a: hz = W_hz.h + b; h3 = h2*silu(z) + h2*Dp ------------
__global__ void k6a_h3(const float* __restrict__ h,
                       const float* __restrict__ Whz,
                       const float* __restrict__ bhz,
                       const float* __restrict__ Dp,
                       float* __restrict__ h3) {
  int idx = blockIdx.x * 256 + threadIdx.x;   // (b*256+o)*64+s
  int s = idx & 63;
  int o = (idx >> 6) & 255;
  int b = idx >> 14;
  const float* hb = h + (long)b * 256 * 64 + s;
  const float* w1 = Whz + (long)o * 256;
  const float* w2 = Whz + (long)(o + 256) * 256;
  float a = bhz[o], z = bhz[o + 256];
  for (int c = 0; c < 256; ++c) {
    float hv = hb[c * 64];
    a += w1[c] * hv;
    z += w2[c] * hv;
  }
  float sig = 1.f / (1.f + __expf(-z));
  h3[idx] = a * (z * sig) + a * Dp[0];
}

// ---------------- K6b: h4 = W_out.h3 + b_out; f32 out + bf16 copy ----------
__global__ void k6b_h4(const float* __restrict__ h3,
                       const float* __restrict__ Wo,
                       const float* __restrict__ bo,
                       unsigned short* __restrict__ h4bf,
                       float* __restrict__ out_tail) {
  int idx = blockIdx.x * 256 + threadIdx.x;
  int s = idx & 63;
  int o = (idx >> 6) & 255;
  int b = idx >> 14;
  const float* hb = h3 + (long)b * 256 * 64 + s;
  const float* w  = Wo + (long)o * 256;
  float a = bo[o];
  for (int c = 0; c < 256; ++c) a += w[c] * hb[c * 64];
  h4bf[idx] = f2bfu(a);
  out_tail[idx] = a;
}

// ---------------- K7: y = h4(256x64) . C_(64xNN), WMMA (K=64) --------------
// One wave per (b, 16-col N tile); the 64x16 C_ panel stays resident (bf16)
// across all 16 M tiles, so C_ is streamed once; h4bf hits L2 as 32B loads.
__global__ void k7_ygemm(const unsigned short* __restrict__ h4bf,
                         const unsigned short* __restrict__ cbf,
                         float* __restrict__ y) {
  const int lane = threadIdx.x & 31;
  const int wid  = threadIdx.x >> 5;
  const int task = blockIdx.x * 8 + wid;   // 16384 = 4 * 4096
  const int b    = task >> 12;
  const int nt   = task & 4095;
  const long n0  = (long)nt * 16;
  const int ncol = lane & 15;
  const int half = lane >> 4;

  v16bf Bt[2];
#pragma unroll
  for (int ks = 0; ks < 2; ++ks) {
    const unsigned short* p =
        cbf + (long)b * SS * NN + (long)(ks * 32 + 16 * half) * NN + n0 + ncol;
    v8u t;
#pragma unroll
    for (int v = 0; v < 8; ++v)
      t[v] = (unsigned)p[(long)(2 * v) * NN] |
             ((unsigned)p[(long)(2 * v + 1) * NN] << 16);
    Bt[ks] = __builtin_bit_cast(v16bf, t);
  }
  for (int ct = 0; ct < 16; ++ct) {
    v8f acc;
#pragma unroll
    for (int j = 0; j < 8; ++j) acc[j] = 0.f;
#pragma unroll
    for (int ks = 0; ks < 2; ++ks) {
      v16bf A = loadfrag_bf(h4bf + ((long)b * 256 + ct * 16 + ncol) * 64 +
                            ks * 32 + 16 * half);
      acc = __builtin_amdgcn_wmma_f32_16x16x32_bf16(false, A, false, Bt[ks],
                                                    (short)0, acc, false, false);
    }
    float* yr = y + ((long)b * 256 + ct * 16) * NN + n0 + ncol;
#pragma unroll
    for (int j = 0; j < 8; ++j)
      yr[(long)(j + 8 * half) * NN] = acc[j];
  }
}

extern "C" void kernel_launch(void* const* d_in, const int* in_sizes, int n_in,
                              void* d_out, int out_size, void* d_ws, size_t ws_size,
                              hipStream_t stream) {
  (void)in_sizes; (void)n_in; (void)out_size; (void)ws_size;
  const float* x    = (const float*)d_in[0];
  const float* Wb   = (const float*)d_in[1];
  const float* bb   = (const float*)d_in[2];
  const float* Wd   = (const float*)d_in[3];
  const float* bd   = (const float*)d_in[4];
  const float* Whz  = (const float*)d_in[5];
  const float* bhz  = (const float*)d_in[6];
  const float* Wo   = (const float*)d_in[7];
  const float* bo   = (const float*)d_in[8];
  // d_in[9] (A): softmax over n is shift-invariant in A[s] -> cancels exactly
  const float* Dp   = (const float*)d_in[10];

  char* ws = (char*)d_ws;
  size_t off = 0;
  float* bcdt_raw      = (float*)(ws + off); off += (size_t)BB * 192 * NN * 4; // 201.3MB
  float* bdt           = (float*)(ws + off); off += (size_t)BB * 128 * NN * 4; // 134.2MB
  unsigned short* cbf  = (unsigned short*)(ws + off); off += (size_t)BB * 64 * NN * 2; // 33.5MB
  unsigned short* ab   = (unsigned short*)(ws + off); off += (size_t)BB * 64 * NN * 2; // 33.5MB
  float* stats         = (float*)(ws + off); off += (size_t)BB * 64 * 2 * 4;
  float* part          = (float*)(ws + off); off += (size_t)64 * BB * 256 * 64 * 4;    // 16MB
  float* h             = (float*)(ws + off); off += (size_t)BB * 256 * 64 * 4;
  float* h3            = (float*)(ws + off); off += (size_t)BB * 256 * 64 * 4;
  unsigned short* h4bf = (unsigned short*)(ws + off); off += (size_t)BB * 256 * 64 * 2;
  unsigned short* wbf  = (unsigned short*)(ws + off); off += (size_t)192 * 256 * 2;

  float* y        = (float*)d_out;
  float* out_tail = (float*)d_out + (size_t)BB * DM * NN;

  k0_wcvt<<<192, 256, 0, stream>>>(Wb, wbf);
  k1_bcdt_gemm<<<2048, 256, 0, stream>>>(x, wbf, bb, bcdt_raw);
  k2_dwconv<<<(BB * 192 * NN) / 256, 256, 0, stream>>>(bcdt_raw, Wd, bd, bdt, cbf);
  k3_softmax_stats<<<BB * 64, 256, 0, stream>>>(bdt, stats);
  k4_ab<<<(BB * 64 * NN) / 256, 256, 0, stream>>>(bdt, stats, ab);
  k5_hgemm<<<512, 256, 0, stream>>>(x, ab, part);
  k5b_reduce<<<256, 256, 0, stream>>>(part, h);
  k6a_h3<<<256, 256, 0, stream>>>(h, Whz, bhz, Dp, h3);
  k6b_h4<<<256, 256, 0, stream>>>(h3, Wo, bo, h4bf, out_tail);
  k7_ygemm<<<2048, 256, 0, stream>>>(h4bf, cbf, y);
}
